// MultiScaleHypercubeLayer_25950192402969
// MI455X (gfx1250) — compile-verified, matrix operand order checked by reading, not run on hardware
//
#include <hip/hip_runtime.h>
#include <math.h>

typedef __attribute__((ext_vector_type(2)))  float    v2f;
typedef __attribute__((ext_vector_type(8)))  float    v8f;
typedef __attribute__((ext_vector_type(16))) _Float16 v16h;

#define D_MODEL 2048
#define HC_DIM 4
#define NVERT 16
#define TOKENS_PER_BLOCK 16
#define THREADS 256
#define N_TOKENS (4 * 4096)

__global__ __launch_bounds__(THREADS)
void hypercube_fused_kernel(const float* __restrict__ x,
                            const float* __restrict__ W_to,
                            const float* __restrict__ W_from,
                            const float* __restrict__ log_temp,
                            const float* __restrict__ scale,
                            float* __restrict__ out)
{
    __shared__ float zs[TOKENS_PER_BLOCK][HC_DIM];
    __shared__ float qs[TOKENS_PER_BLOCK][HC_DIM];

    const int tid       = threadIdx.x;
    const int tokenBase = blockIdx.x * TOKENS_PER_BLOCK;

    // ---------------- Phase 1: z[t][k] = dot(x[t], W_to[k]) ----------------
    // 16 lanes per token, float4 streaming loads (fully coalesced).
    {
        const int   tLocal = tid >> 4;
        const int   lane16 = tid & 15;
        const float* xrow  = x + (size_t)(tokenBase + tLocal) * D_MODEL;

        float acc0 = 0.f, acc1 = 0.f, acc2 = 0.f, acc3 = 0.f;
        #pragma unroll 4
        for (int j = 0; j < D_MODEL / (16 * 4); ++j) {
            const int dim = (j * 16 + lane16) * 4;
            float4 xv = *(const float4*)(xrow + dim);
            float4 w0 = *(const float4*)(W_to + 0 * D_MODEL + dim);
            float4 w1 = *(const float4*)(W_to + 1 * D_MODEL + dim);
            float4 w2 = *(const float4*)(W_to + 2 * D_MODEL + dim);
            float4 w3 = *(const float4*)(W_to + 3 * D_MODEL + dim);
            acc0 += xv.x * w0.x + xv.y * w0.y + xv.z * w0.z + xv.w * w0.w;
            acc1 += xv.x * w1.x + xv.y * w1.y + xv.z * w1.z + xv.w * w1.w;
            acc2 += xv.x * w2.x + xv.y * w2.y + xv.z * w2.z + xv.w * w2.w;
            acc3 += xv.x * w3.x + xv.y * w3.y + xv.z * w3.z + xv.w * w3.w;
        }
        // reduce within each 16-lane half-wave (wave32, width=16 keeps tokens separate)
        #pragma unroll
        for (int off = 8; off > 0; off >>= 1) {
            acc0 += __shfl_xor(acc0, off, 16);
            acc1 += __shfl_xor(acc1, off, 16);
            acc2 += __shfl_xor(acc2, off, 16);
            acc3 += __shfl_xor(acc3, off, 16);
        }
        if (lane16 == 0) {
            zs[tLocal][0] = acc0; zs[tLocal][1] = acc1;
            zs[tLocal][2] = acc2; zs[tLocal][3] = acc3;
        }
    }
    __syncthreads();

    // ---------------- Phase 2: softmax over 16 hypercube vertices ----------
    if (tid < TOKENS_PER_BLOCK) {
        const float z0 = zs[tid][0], z1 = zs[tid][1];
        const float z2 = zs[tid][2], z3 = zs[tid][3];
        float temp = __expf(log_temp[0]);
        temp = fminf(fmaxf(temp, 0.01f), 5.0f);
        const float sc = scale[0];

        float s[NVERT];
        float mx = -3.0e38f;
        #pragma unroll
        for (int k = 0; k < NVERT; ++k) {
            // itertools.product([0,1],repeat=4): v[k][j] = bit (3-j) of k
            const float b0 = (float)((k >> 3) & 1);
            const float b1 = (float)((k >> 2) & 1);
            const float b2 = (float)((k >> 1) & 1);
            const float b3 = (float)( k       & 1);
            const float d0 = z0 - b0, d1 = z1 - b1, d2 = z2 - b2, d3 = z3 - b3;
            const float dist = sqrtf(d0*d0 + d1*d1 + d2*d2 + d3*d3);
            const float sk = -dist / temp;
            s[k] = sk;
            mx = fmaxf(mx, sk);
        }
        float sum = 0.f, q0 = 0.f, q1 = 0.f, q2 = 0.f, q3 = 0.f;
        #pragma unroll
        for (int k = 0; k < NVERT; ++k) {
            const float w = __expf(s[k] - mx);
            sum += w;
            q0 += w * (float)((k >> 3) & 1);
            q1 += w * (float)((k >> 2) & 1);
            q2 += w * (float)((k >> 1) & 1);
            q3 += w * (float)( k       & 1);
        }
        const float inv = sc / sum;          // fold 'scale' into q
        qs[tid][0] = q0 * inv; qs[tid][1] = q1 * inv;
        qs[tid][2] = q2 * inv; qs[tid][3] = q3 * inv;
    }
    __syncthreads();

    // ---------------- Phase 3: out = x + (scale*q) @ W_from^T via WMMA -----
    // D(16x16) = A(16x4) * B(4x16) + C(16x16), one WMMA per 16-token x 16-dim tile.
    {
        const int wave   = tid >> 5;     // 0..7
        const int l      = tid & 31;
        const int half   = l >> 4;       // 0/1 (lane>=16)
        const int lane16 = l & 15;

#if __has_builtin(__builtin_amdgcn_wmma_f32_16x16x4_f32)
        // A layout (32-bit A 16x4): lane M = l%16, VGPR r holds K = r + 2*half
        v2f A;
        A[0] = qs[lane16][2 * half + 0];
        A[1] = qs[lane16][2 * half + 1];
#else
        // Fallback: f16 WMMA with K=32 zero-padded (A 16-bit 16x32 layout:
        // lanes 0-15 elements 0..7 hold K=0..7; lanes 16-31 hold K=8..15 -> zero)
        const float am = (half == 0) ? 1.0f : 0.0f;
        v16h Ah;
        #pragma unroll
        for (int e = 0; e < 16; ++e) Ah[e] = (_Float16)0.0f;
        Ah[0] = (_Float16)(qs[lane16][0] * am);
        Ah[1] = (_Float16)(qs[lane16][1] * am);
        Ah[2] = (_Float16)(qs[lane16][2] * am);
        Ah[3] = (_Float16)(qs[lane16][3] * am);
#endif

        const float* xbase = x   + (size_t)tokenBase * D_MODEL;
        float*       obase = out + (size_t)tokenBase * D_MODEL;
        const int    dim0  = wave * (D_MODEL / 8);   // 256 dims per wave

        #pragma unroll
        for (int c = 0; c < (D_MODEL / 8) / 16; ++c) {
            const int dim = dim0 + c * 16 + lane16;

            // C layout (32-bit C/D 16x16): VGPR r -> row M = r + 8*half, col N = lane16
            v8f C;
            #pragma unroll
            for (int r = 0; r < 8; ++r)
                C[r] = xbase[(size_t)(r + 8 * half) * D_MODEL + dim];

            v8f Dm;
#if __has_builtin(__builtin_amdgcn_wmma_f32_16x16x4_f32)
            // B layout (32-bit B 4x16): lane N = l%16, VGPR r holds K = r + 2*half
            v2f B;
            B[0] = W_from[dim * HC_DIM + 2 * half + 0];
            B[1] = W_from[dim * HC_DIM + 2 * half + 1];
            Dm = __builtin_amdgcn_wmma_f32_16x16x4_f32(
                     false, A, false, B, (short)0, C, false, false);
#else
            // B 16-bit 32x16: lanes 0-15 elements 0..3 hold K=0..3 (same as A), all other elements -> zero
            const float bm = (half == 0) ? 1.0f : 0.0f;
            v16h Bh;
            #pragma unroll
            for (int e = 0; e < 16; ++e) Bh[e] = (_Float16)0.0f;
            Bh[0] = (_Float16)(W_from[dim * HC_DIM + 0] * bm);
            Bh[1] = (_Float16)(W_from[dim * HC_DIM + 1] * bm);
            Bh[2] = (_Float16)(W_from[dim * HC_DIM + 2] * bm);
            Bh[3] = (_Float16)(W_from[dim * HC_DIM + 3] * bm);
            Dm = __builtin_amdgcn_wmma_f32_16x16x32_f16(
                     false, Ah, false, Bh, (short)0, C, false, false);
#endif
            #pragma unroll
            for (int r = 0; r < 8; ++r)
                obase[(size_t)(r + 8 * half) * D_MODEL + dim] = Dm[r];
        }
    }
}

extern "C" void kernel_launch(void* const* d_in, const int* in_sizes, int n_in,
                              void* d_out, int out_size, void* d_ws, size_t ws_size,
                              hipStream_t stream) {
    const float* x        = (const float*)d_in[0];
    const float* W_to     = (const float*)d_in[1];
    const float* W_from   = (const float*)d_in[2];
    const float* log_temp = (const float*)d_in[3];
    const float* scale    = (const float*)d_in[4];
    float*       out      = (float*)d_out;

    const int nBlocks = N_TOKENS / TOKENS_PER_BLOCK;   // 1024
    hipLaunchKernelGGL(hypercube_fused_kernel,
                       dim3(nBlocks), dim3(THREADS), 0, stream,
                       x, W_to, W_from, log_temp, scale, out);
}
